// SMambaResidualDenseBlock_out_24670292149090
// MI455X (gfx1250) — compile-verified, matrix operand order checked by reading, not run on hardware
//
#include <hip/hip_runtime.h>
#include <hip/hip_bf16.h>

typedef __bf16 bf16;
typedef __attribute__((ext_vector_type(16))) __bf16 v16bf;
typedef __attribute__((ext_vector_type(8)))  float  v8f;
typedef __attribute__((ext_vector_type(4)))  int    i32x4;
typedef __attribute__((ext_vector_type(8)))  int    i32x8;

#define LPIX 4096
#define BATCH 2
#define DM 192
#define DI 384
#define NST 16
#define DTR 12
#define KD 4

// ---------------- workspace layout (bytes) ----------------
static constexpr size_t OFF_F    = 0;                         // f  : B*192*L fp32     6291456
static constexpr size_t OFF_XN   = OFF_F    + 6291456;        // xn : B*L*192 bf16     3145728
static constexpr size_t OFF_XP   = OFF_XN   + 3145728;        // xp : B*384*L bf16     6291456
static constexpr size_t OFF_Z    = OFF_XP   + 6291456;        // z  : B*L*384 bf16     6291456
static constexpr size_t OFF_XC   = OFF_Z    + 6291456;        // xc : B*384*L bf16     6291456
static constexpr size_t OFF_XDBL = OFF_XC   + 6291456;        // xdbl: B*4*48*L fp32   6291456
static constexpr size_t OFF_Y    = OFF_XDBL + 6291456;        // y  : B*L*384 fp32    12582912
static constexpr size_t OFF_YB   = OFF_Y    + 12582912;       // yb : B*L*384 bf16     6291456
static constexpr size_t OFF_SA   = OFF_YB   + 6291456;        // sa : B*192*L fp32     6291456
static constexpr size_t OFF_WIN  = OFF_SA   + 6291456;        // in_proj bf16           294912
static constexpr size_t OFF_WXP  = OFF_WIN  + 294912;         // x_proj bf16 (pad 48)   147456
static constexpr size_t OFF_WOUT = OFF_WXP  + 147456;         // out_proj bf16          147456

// ---------------- WMMA helpers (ISA 7.12.2 layouts) ----------------
__device__ __forceinline__ int lane_id() { return threadIdx.x & 31; }

__device__ __forceinline__ v8f vzero() {
  v8f v;
#pragma unroll
  for (int i = 0; i < 8; ++i) v[i] = 0.f;
  return v;
}

// Load a 16x32 bf16 fragment from a row-major matrix (16 rows of `ldm` elems).
// Works for both A (rows = M) and B (rows = N, K contiguous per row).
__device__ __forceinline__ v16bf load_frag_rowmajor(const bf16* base, int ldm, int k0) {
  int l = lane_id();
  int r = l & 15;
  int kh = l >> 4;
  const bf16* p = base + (size_t)r * ldm + k0 + kh * 8;
  v16bf f;
#pragma unroll
  for (int e = 0; e < 16; ++e) f[e] = p[(e < 8) ? e : (e + 8)];
  return f;
}

__device__ __forceinline__ v8f wmma_bf16(v16bf a, v16bf b, v8f c) {
  return __builtin_amdgcn_wmma_f32_16x16x32_bf16(false, a, false, b, (short)0, c,
                                                 false, false);
}

// ---------------- Tensor Data Mover: 1-D contiguous global -> LDS ----------------
// Builds a D# (ISA ch.8): count=1, type=2 (image), data_size=8B units, single row
// tile of `bytes/8` units. Groups 2/3 passed as zero-filled quads (2-D mode).
__device__ __forceinline__ void tdm_copy_1d(const void* gsrc, void* lds_dst, int bytes) {
  unsigned units = (unsigned)(bytes >> 3);
  unsigned lds_addr = (unsigned)(uintptr_t)lds_dst;  // addr[31:0] = LDS offset
  unsigned long long ga = (unsigned long long)(uintptr_t)gsrc;
  i32x4 g0;
  i32x8 g1;
  i32x4 gz;
  g0[0] = 1;                                            // count=1 (valid user D#)
  g0[1] = (int)lds_addr;                                // lds_addr
  g0[2] = (int)(unsigned)ga;                            // global_addr[31:0]
  g0[3] = (int)(((unsigned)(ga >> 32) & 0x01FFFFFFu) | 0x80000000u);  // ga[56:32] | type=2
  g1[0] = 0x00030000;                                   // data_size=3 (8B), mask=0
  g1[1] = (int)((units & 0xFFFFu) << 16);               // tensor_dim0[15:0]
  g1[2] = (int)((units >> 16) | (1u << 16));            // tensor_dim0[31:16] | tensor_dim1=1
  g1[3] = (int)((units & 0xFFFFu) << 16);               // tile_dim0 = units
  g1[4] = 0;                                            // tile_dim1/2 unused
  g1[5] = (int)units;                                   // tensor_dim0_stride lo
  g1[6] = (int)((units >> 16) & 0xFFFFu);               // stride hi
  g1[7] = 0;
  gz[0] = 0; gz[1] = 0; gz[2] = 0; gz[3] = 0;
  asm volatile("tensor_load_to_lds %0, %1, %2, %3"
               :
               : "s"(g0), "s"(g1), "s"(gz), "s"(gz)
               : "memory");
}

// scan-direction index map: scan position l -> spatial position
__device__ __forceinline__ int dirmap(int k, int l) {
  if (k & 2) l = LPIX - 1 - l;
  if (k & 1) { int w = l >> 6, h = l & 63; l = (h << 6) | w; }
  return l;
}

// ---------------- kernels ----------------
__global__ void prep_weights_k(const float* win_f, const float* wxp_f, const float* wout_f,
                               bf16* win, bf16* wxp, bf16* wout) {
  int t = blockIdx.x * blockDim.x + threadIdx.x;
  const int NIN = 768 * DM, NXP = KD * 48 * DI, NOUT = DM * DI;
  if (t < NIN) {
    win[t] = (bf16)win_f[t];
  } else if (t < NIN + NXP) {
    int i = t - NIN;
    int d = i % DI, c = (i / DI) % 48, k = i / (DI * 48);
    wxp[i] = (c < 44) ? (bf16)wxp_f[((k * 44) + c) * DI + d] : (bf16)0.f;
  } else if (t < NIN + NXP + NOUT) {
    int i = t - NIN - NXP;
    wout[i] = (bf16)wout_f[i];
  }
}

__global__ void copy_x_k(const float* x, float* f) {
  int t = blockIdx.x * blockDim.x + threadIdx.x;
  if (t >= BATCH * 64 * LPIX) return;
  int p = t % LPIX, c = (t / LPIX) % 64, b = t / (LPIX * 64);
  f[(((size_t)b * DM) + c) * LPIX + p] = x[t];
}

// one dense-block conv stage: reads f[:, :cin], writes lrelu result at f[:, cin:cin+32]
__global__ void dense_conv_k(float* f, const float* w, const float* bias, int cin) {
  int t = blockIdx.x * blockDim.x + threadIdx.x;
  if (t >= BATCH * 32 * LPIX) return;
  int p = t % LPIX, oc = (t / LPIX) % 32, b = t / (LPIX * 32);
  int h = p >> 6, wc = p & 63;
  float acc = bias[oc];
  const float* wrow = w + (size_t)oc * cin * 9;
  const float* fb = f + (size_t)b * DM * LPIX;
  for (int ci = 0; ci < cin; ++ci) {
    const float* fc = fb + (size_t)ci * LPIX;
    const float* wp = wrow + ci * 9;
    __builtin_prefetch(fc + LPIX, 0, 1);
#pragma unroll
    for (int ky = 0; ky < 3; ++ky) {
      int ih = h + ky - 1;
      if ((unsigned)ih >= 64u) continue;
#pragma unroll
      for (int kx = 0; kx < 3; ++kx) {
        int iw = wc + kx - 1;
        if ((unsigned)iw >= 64u) continue;
        acc = fmaf(wp[ky * 3 + kx], fc[ih * 64 + iw], acc);
      }
    }
  }
  float v = acc > 0.f ? acc : 0.01f * acc;
  f[(((size_t)b * DM) + cin + oc) * LPIX + p] = v;
}

__global__ void ln_k(const float* f, const float* g, const float* be, bf16* xn) {
  int t = blockIdx.x * blockDim.x + threadIdx.x;
  if (t >= BATCH * LPIX) return;
  int p = t % LPIX, b = t / LPIX;
  const float* fb = f + (size_t)b * DM * LPIX + p;
  float s = 0.f, s2 = 0.f;
  for (int c = 0; c < DM; ++c) { float v = fb[(size_t)c * LPIX]; s += v; s2 += v * v; }
  float mu = s / DM;
  float rs = rsqrtf(s2 / DM - mu * mu + 1e-5f);
  bf16* o = xn + (size_t)t * DM;
  for (int c = 0; c < DM; ++c) {
    float v = fb[(size_t)c * LPIX];
    o[c] = (bf16)(((v - mu) * rs) * g[c] + be[c]);
  }
}

// xz = xn @ in_proj_w^T : M = B*L, N = 768, K = 192.
// Block tile: 64 pixels x 192 outputs. A tile (24 KB) and weight slab (72 KB) are
// DMA'd into LDS by the Tensor Data Mover, then 8 waves consume ds_load fragments.
__global__ void __launch_bounds__(256) inproj_k(const bf16* xn, const bf16* win,
                                                bf16* xp, bf16* z) {
  __shared__ __align__(16) bf16 Asl[64 * DM];    // 24576 B
  __shared__ __align__(16) bf16 Bsl[192 * DM];   // 73728 B
  int mgrp = blockIdx.x & 127;   // 128 groups of 64 pixels
  int slab = blockIdx.x >> 7;    // 4 slabs of 192 outputs
  int m0 = mgrp * 64;
  int n0 = slab * 192;

  int wv = threadIdx.x >> 5;
  if (wv == 0) {
    tdm_copy_1d(xn + (size_t)m0 * DM, Asl, 64 * DM * 2);
    tdm_copy_1d(win + (size_t)n0 * DM, Bsl, 192 * DM * 2);
    __builtin_amdgcn_s_wait_tensorcnt(0);
  }
  __syncthreads();

  int mt = wv >> 1;              // 4 M-tiles
  int half = wv & 1;             // 2 halves x 6 N-tiles = 12 N-tiles (192 outs)
  const bf16* arow = Asl + (size_t)mt * 16 * DM;

  v8f acc[6];
#pragma unroll
  for (int j = 0; j < 6; ++j) acc[j] = vzero();
  for (int k0 = 0; k0 < DM; k0 += 32) {
    v16bf a = load_frag_rowmajor(arow, DM, k0);
#pragma unroll
    for (int j = 0; j < 6; ++j) {
      int ntl = half * 6 + j;
      v16bf bf = load_frag_rowmajor(Bsl + (size_t)ntl * 16 * DM, DM, k0);
      acc[j] = wmma_bf16(a, bf, acc[j]);
    }
  }

  int l = lane_id();
  int ncol = l & 15, kh = l >> 4;
  int gbase = m0 + mt * 16;
  int b = gbase / LPIX;          // 64 | LPIX, tile stays inside one batch
#pragma unroll
  for (int j = 0; j < 6; ++j) {
    int n = n0 + (half * 6 + j) * 16 + ncol;
#pragma unroll
    for (int r = 0; r < 8; ++r) {
      int g = gbase + r + 8 * kh;
      int p = g % LPIX;
      float v = acc[j][r];
      if (n < DI) xp[(((size_t)b * DI) + n) * LPIX + p] = (bf16)v;
      else        z[(size_t)g * DI + (n - DI)] = (bf16)v;
    }
  }
}

__global__ void dwconv_silu_k(const bf16* xp, const float* w, const float* bias, bf16* xc) {
  int t = blockIdx.x * blockDim.x + threadIdx.x;
  if (t >= BATCH * DI * LPIX) return;
  int p = t % LPIX, d = (t / LPIX) % DI, b = t / (LPIX * DI);
  int h = p >> 6, wc = p & 63;
  float acc = bias[d];
  const bf16* src = xp + (((size_t)b * DI) + d) * LPIX;
  const float* wp = w + d * 9;
#pragma unroll
  for (int ky = 0; ky < 3; ++ky) {
    int ih = h + ky - 1;
    if ((unsigned)ih >= 64u) continue;
#pragma unroll
    for (int kx = 0; kx < 3; ++kx) {
      int iw = wc + kx - 1;
      if ((unsigned)iw >= 64u) continue;
      acc = fmaf(wp[ky * 3 + kx], (float)src[ih * 64 + iw], acc);
    }
  }
  float s = acc / (1.f + __expf(-acc));
  xc[t] = (bf16)s;
}

// x_dbl[b,k,c,l] = sum_d x_proj_w[k,c,d] * xs[b,k,d,l]; per-direction gather into LDS
__global__ void __launch_bounds__(128) xproj_k(const bf16* xc, const bf16* wxp, float* xdbl) {
  __shared__ bf16 As[16 * 392];  // padded stride vs 64-bank LDS
  int mt = blockIdx.x & 255;
  int k  = (blockIdx.x >> 8) & 3;
  int b  = blockIdx.x >> 10;
  for (int i = threadIdx.x; i < 16 * DI; i += 128) {
    int m = i / DI, d = i % DI;
    int sp = dirmap(k, mt * 16 + m);
    As[m * 392 + d] = xc[(((size_t)b * DI) + d) * LPIX + sp];
  }
  __syncthreads();
  int wv = threadIdx.x >> 5;
  if (wv < 3) {  // 3 N-tiles cover 48 (44 real) output channels
    v8f acc = vzero();
    for (int k0 = 0; k0 < DI; k0 += 32) {
      v16bf a  = load_frag_rowmajor(As, 392, k0);
      v16bf bf = load_frag_rowmajor(wxp + (((size_t)k * 48) + wv * 16) * DI, DI, k0);
      acc = wmma_bf16(a, bf, acc);
    }
    int l = lane_id();
    int c = wv * 16 + (l & 15), kh = l >> 4;
    float* out = xdbl + (((size_t)(b * KD + k)) * 48 + c) * LPIX + mt * 16;
#pragma unroll
    for (int r = 0; r < 8; ++r) out[r + 8 * kh] = acc[r];
  }
}

// sequential selective scan; dt computed on the fly; 4 directions merged via f32 atomics
__global__ void __launch_bounds__(128) scan_k(const float* xdbl, const bf16* xc,
                                              const float* dt_w, const float* dt_b,
                                              const float* A_logs, const float* Ds,
                                              float* ybuf) {
  int seg = blockIdx.x % 3;
  int bk = blockIdx.x / 3;
  int k = bk & 3, b = bk >> 2;
  int d = seg * 128 + threadIdx.x;

  float dtw[DTR];
  const float* dwp = dt_w + ((size_t)k * DI + d) * DTR;
#pragma unroll
  for (int r = 0; r < DTR; ++r) dtw[r] = dwp[r];
  float dtb = dt_b[k * DI + d];
  float Ar[NST];
  const float* ap = A_logs + ((size_t)k * DI + d) * NST;
#pragma unroll
  for (int n = 0; n < NST; ++n) Ar[n] = -__expf(ap[n]);
  float Dv = Ds[k * DI + d];
  float hst[NST];
#pragma unroll
  for (int n = 0; n < NST; ++n) hst[n] = 0.f;

  __shared__ float col[44];
  const float* xd = xdbl + ((size_t)(b * KD + k)) * 48 * LPIX;
  const bf16* xcb = xc + (((size_t)b * DI) + d) * LPIX;
  float* yb = ybuf + (size_t)b * LPIX * DI + d;

  for (int l = 0; l < LPIX; ++l) {
    __syncthreads();
    if (threadIdx.x < 44) col[threadIdx.x] = xd[(size_t)threadIdx.x * LPIX + l];
    __syncthreads();
    int sp = dirmap(k, l);
    float u = (float)xcb[sp];
    float dt = dtb;
#pragma unroll
    for (int r = 0; r < DTR; ++r) dt = fmaf(dtw[r], col[r], dt);
    float delta = (dt > 20.f) ? dt : __logf(1.f + __expf(dt));
    float du = delta * u;
    float y = 0.f;
#pragma unroll
    for (int n = 0; n < NST; ++n) {
      hst[n] = __expf(delta * Ar[n]) * hst[n] + du * col[12 + n];
      y = fmaf(hst[n], col[28 + n], y);
    }
    unsafeAtomicAdd(yb + (size_t)sp * DI, y + Dv * u);
  }
}

__global__ void postnorm_gate_k(const float* ybuf, const bf16* z, const float* g,
                                const float* be, bf16* yb16) {
  int t = blockIdx.x * blockDim.x + threadIdx.x;
  if (t >= BATCH * LPIX) return;
  const float* yp = ybuf + (size_t)t * DI;
  float s = 0.f, s2 = 0.f;
  for (int c = 0; c < DI; ++c) { float v = yp[c]; s += v; s2 += v * v; }
  float mu = s / DI;
  float rs = rsqrtf(s2 / DI - mu * mu + 1e-5f);
  const bf16* zp = z + (size_t)t * DI;
  bf16* op = yb16 + (size_t)t * DI;
  for (int c = 0; c < DI; ++c) {
    float v = (yp[c] - mu) * rs * g[c] + be[c];
    float zv = (float)zp[c];
    float sz = zv / (1.f + __expf(-zv));
    op[c] = (bf16)(v * sz);
  }
}

// sa = f + yb16 @ out_proj_w^T  (stored NCHW for the final conv)
__global__ void __launch_bounds__(128) outproj_k(const bf16* yb16, const bf16* wout,
                                                 const float* f, float* sa) {
  int mt = blockIdx.x;
  int wv = threadIdx.x >> 5;  // 4 waves * 3 N-tiles = 12 tiles (192 outs)
  const bf16* arow = yb16 + (size_t)mt * 16 * DI;
  int gbase = mt * 16;
  int b = gbase / LPIX;
#pragma unroll
  for (int j = 0; j < 3; ++j) {
    int nt = wv * 3 + j;
    v8f acc = vzero();
    for (int k0 = 0; k0 < DI; k0 += 32) {
      v16bf a  = load_frag_rowmajor(arow, DI, k0);
      v16bf bf = load_frag_rowmajor(wout + (size_t)nt * 16 * DI, DI, k0);
      acc = wmma_bf16(a, bf, acc);
    }
    int l = lane_id();
    int c = nt * 16 + (l & 15), kh = l >> 4;
#pragma unroll
    for (int r = 0; r < 8; ++r) {
      int p = (gbase % LPIX) + r + 8 * kh;
      size_t idx = (((size_t)b * DM) + c) * LPIX + p;
      sa[idx] = acc[r] + f[idx];
    }
  }
}

__global__ void c5_k(const float* sa, const float* w, const float* bias, float* out) {
  int t = blockIdx.x * blockDim.x + threadIdx.x;
  if (t >= BATCH * 64 * LPIX) return;
  int p = t % LPIX, oc = (t / LPIX) % 64, b = t / (LPIX * 64);
  int h = p >> 6, wc = p & 63;
  float acc = bias[oc];
  const float* wrow = w + (size_t)oc * DM * 9;
  const float* sb = sa + (size_t)b * DM * LPIX;
  for (int ci = 0; ci < DM; ++ci) {
    const float* fc = sb + (size_t)ci * LPIX;
    const float* wp = wrow + ci * 9;
    __builtin_prefetch(fc + LPIX, 0, 1);
#pragma unroll
    for (int ky = 0; ky < 3; ++ky) {
      int ih = h + ky - 1;
      if ((unsigned)ih >= 64u) continue;
#pragma unroll
      for (int kx = 0; kx < 3; ++kx) {
        int iw = wc + kx - 1;
        if ((unsigned)iw >= 64u) continue;
        acc = fmaf(wp[ky * 3 + kx], fc[ih * 64 + iw], acc);
      }
    }
  }
  out[t] = acc;
}

// ---------------- host launcher ----------------
extern "C" void kernel_launch(void* const* d_in, const int* in_sizes, int n_in,
                              void* d_out, int out_size, void* d_ws, size_t ws_size,
                              hipStream_t stream) {
  const float* x    = (const float*)d_in[0];
  const float* c1w  = (const float*)d_in[1];
  const float* c1b  = (const float*)d_in[2];
  const float* c2w  = (const float*)d_in[3];
  const float* c2b  = (const float*)d_in[4];
  const float* c3w  = (const float*)d_in[5];
  const float* c3b  = (const float*)d_in[6];
  const float* c4w  = (const float*)d_in[7];
  const float* c4b  = (const float*)d_in[8];
  const float* c5w  = (const float*)d_in[9];
  const float* c5b  = (const float*)d_in[10];
  const float* ln1g = (const float*)d_in[11];
  const float* ln1b = (const float*)d_in[12];
  const float* wInF = (const float*)d_in[13];
  const float* dww  = (const float*)d_in[14];
  const float* dwb  = (const float*)d_in[15];
  const float* wXpF = (const float*)d_in[16];
  const float* dtw  = (const float*)d_in[17];
  const float* dtb  = (const float*)d_in[18];
  const float* Alog = (const float*)d_in[19];
  const float* Ds   = (const float*)d_in[20];
  const float* ong  = (const float*)d_in[21];
  const float* onb  = (const float*)d_in[22];
  const float* wOutF= (const float*)d_in[23];

  char* ws = (char*)d_ws;
  float* F    = (float*)(ws + OFF_F);
  bf16*  XN   = (bf16*)(ws + OFF_XN);
  bf16*  XP   = (bf16*)(ws + OFF_XP);
  bf16*  Z    = (bf16*)(ws + OFF_Z);
  bf16*  XC   = (bf16*)(ws + OFF_XC);
  float* XDBL = (float*)(ws + OFF_XDBL);
  float* Y    = (float*)(ws + OFF_Y);
  bf16*  YB   = (bf16*)(ws + OFF_YB);
  float* SA   = (float*)(ws + OFF_SA);
  bf16*  WIN  = (bf16*)(ws + OFF_WIN);
  bf16*  WXP  = (bf16*)(ws + OFF_WXP);
  bf16*  WOUT = (bf16*)(ws + OFF_WOUT);

  hipMemsetAsync(ws + OFF_Y, 0, (size_t)BATCH * LPIX * DI * sizeof(float), stream);

  prep_weights_k<<<(768 * DM + KD * 48 * DI + DM * DI + 255) / 256, 256, 0, stream>>>(
      wInF, wXpF, wOutF, WIN, WXP, WOUT);
  copy_x_k<<<(BATCH * 64 * LPIX + 255) / 256, 256, 0, stream>>>(x, F);
  dense_conv_k<<<(BATCH * 32 * LPIX + 255) / 256, 256, 0, stream>>>(F, c1w, c1b, 64);
  dense_conv_k<<<(BATCH * 32 * LPIX + 255) / 256, 256, 0, stream>>>(F, c2w, c2b, 96);
  dense_conv_k<<<(BATCH * 32 * LPIX + 255) / 256, 256, 0, stream>>>(F, c3w, c3b, 128);
  dense_conv_k<<<(BATCH * 32 * LPIX + 255) / 256, 256, 0, stream>>>(F, c4w, c4b, 160);
  ln_k<<<(BATCH * LPIX + 255) / 256, 256, 0, stream>>>(F, ln1g, ln1b, XN);
  inproj_k<<<(BATCH * LPIX / 64) * 4, 256, 0, stream>>>(XN, WIN, XP, Z);
  dwconv_silu_k<<<(BATCH * DI * LPIX + 255) / 256, 256, 0, stream>>>(XP, dww, dwb, XC);
  xproj_k<<<BATCH * KD * (LPIX / 16), 128, 0, stream>>>(XC, WXP, XDBL);
  scan_k<<<BATCH * KD * 3, 128, 0, stream>>>(XDBL, XC, dtw, dtb, Alog, Ds, Y);
  postnorm_gate_k<<<(BATCH * LPIX + 255) / 256, 256, 0, stream>>>(Y, Z, ong, onb, YB);
  outproj_k<<<BATCH * LPIX / 16, 128, 0, stream>>>(YB, WOUT, F, SA);
  c5_k<<<(BATCH * 64 * LPIX + 255) / 256, 256, 0, stream>>>(SA, c5w, c5b, (float*)d_out);
}
